// MultiHeadSelfAttention_59742995087739
// MI455X (gfx1250) — matrix operand order repaired and checked by reading, not compiled, not run on hardware
//
#include <hip/hip_runtime.h>
#include <hip/hip_bf16.h>

// ---------------------------------------------------------------------------
// Problem constants (from reference)
// ---------------------------------------------------------------------------
#define D_MODEL   1024
#define NUM_HEADS 16
#define D_HEAD    64
#define BATCH     2
#define SEQ       2048
#define M_TOTAL   (BATCH * SEQ)   // 4096 tokens

// ---------------------------------------------------------------------------
// WMMA fragment types (gfx1250, wave32): 16 bf16 per lane = 8 VGPRs
// ---------------------------------------------------------------------------
typedef __attribute__((ext_vector_type(16))) __bf16        v16bf;
typedef __attribute__((ext_vector_type(8)))  float         v8f;
typedef __attribute__((ext_vector_type(4)))  unsigned int  uint4v;
typedef __attribute__((ext_vector_type(4)))  unsigned int  v4u;
typedef __attribute__((ext_vector_type(8)))  int           v8i;
typedef __attribute__((ext_vector_type(4)))  int           v4i;

union Frag {
    v16bf        b;
    uint4v       u4[2];
    unsigned int u[8];
};

__device__ __forceinline__ unsigned short f2bf(float f) {
    unsigned int u = __builtin_bit_cast(unsigned int, f);
    unsigned int r = u + 0x7FFFu + ((u >> 16) & 1u);   // round-to-nearest-even
    return (unsigned short)(r >> 16);
}

__device__ __forceinline__ v8f wmma_bf16(const Frag& a, const Frag& b, v8f c) {
    // D = A(16x32 bf16) * B(32x16 bf16) + C(16x16 f32)
    return __builtin_amdgcn_wmma_f32_16x16x32_bf16(
        /*neg_a=*/false, a.b, /*neg_b=*/false, b.b,
        /*c_mod=*/(short)0, c, /*reuse_a=*/false, /*reuse_b=*/false);
}

// A-fragment (M=16 x K=32, 16-bit): lane L holds row M = L%16;
//   element i -> k = i + 16*(L/16)
// => one contiguous 32B run per lane at col_base + 16*(L/16).
__device__ __forceinline__ Frag load_afrag(const unsigned short* p, int ld,
                                           int row_base, int col_base) {
    const int lane = threadIdx.x & 31;
    const unsigned short* q =
        p + (size_t)(row_base + (lane & 15)) * ld + col_base + ((lane >> 4) << 4);
    Frag f;
    f.u4[0] = *(const uint4v*)(q);
    f.u4[1] = *(const uint4v*)(q + 8);
    return f;
}

// B-fragment (K=32 x N=16, 16-bit): lane L holds col N = L%16;
//   element i -> k = i + 16*(L/16)
// Source stored row-major with the *N index as the row* and k contiguous:
// one contiguous 32B run per lane.
__device__ __forceinline__ Frag load_bfrag(const unsigned short* p, int ld,
                                           int row_base, int col_base) {
    const int lane = threadIdx.x & 31;
    const unsigned short* q =
        p + (size_t)(row_base + (lane & 15)) * ld + col_base + ((lane >> 4) << 4);
    Frag f;
    f.u4[0] = *(const uint4v*)(q);
    f.u4[1] = *(const uint4v*)(q + 8);
    return f;
}

// B-fragment from a TDM-staged LDS tile: 32 bf16 per row, padded to an
// 80-byte (40-element) row stride by the TDM pad feature (bank-conflict-free,
// 16B aligned).  Lane L reads row (row_base + L%16), 32B at element 16*(L/16).
#define LDS_ROW_ELEMS 40   // 64B data + 16B pad
__device__ __forceinline__ Frag load_bfrag_lds(const unsigned short* tile,
                                               int row_base) {
    const int lane = threadIdx.x & 31;
    const unsigned short* q =
        tile + (row_base + (lane & 15)) * LDS_ROW_ELEMS + ((lane >> 4) << 4);
    Frag f;
    f.u4[0] = *(const uint4v*)(q);
    f.u4[1] = *(const uint4v*)(q + 8);
    return f;
}

// ---------------------------------------------------------------------------
// Tensor Data Mover: DMA a 2D bf16 tile (ncols contiguous elems x nrows,
// row stride = stride_elems) from global memory into LDS at lds_off.
// LDS destination is padded +4 DWORDs every 16 DWORDs (pad_interval=3,
// pad_amount=3) => 80B per 64B tile row.
// D# layout per CDNA5 ISA ch.8 (group0/group1; groups 2-3 zero for 2D).
// ---------------------------------------------------------------------------
__device__ __forceinline__ void tdm_load_2d(unsigned lds_off, const void* gsrc,
                                            unsigned ncols, unsigned nrows,
                                            unsigned stride_elems) {
    const unsigned long long ga = (unsigned long long)(size_t)gsrc;
    v4u g0;
    g0[0] = 1u;                                           // count=1 (valid), user mode
    g0[1] = lds_off;                                      // lds_addr [63:32]
    g0[2] = (unsigned)ga;                                 // global_addr [95:64]
    g0[3] = (unsigned)((ga >> 32) & 0x01FFFFFFu) | 0x80000000u; // addr[56:32] | type=2

    v8i g1;
    g1[0] = (int)((1u << 16)            // data_size = 1 -> 2 bytes
                | (1u << 20)            // pad_enable
                | (3u << 22)            // pad_interval: 16 DWORDs (64B)
                | (3u << 25));          // pad_amount: 4 DWORDs (16B)
    g1[1] = (int)((ncols & 0xFFFFu) << 16);                         // tensor_dim0 lo
    g1[2] = (int)(((ncols >> 16) & 0xFFFFu) | ((nrows & 0xFFFFu) << 16)); // d0 hi | d1 lo
    g1[3] = (int)(((nrows >> 16) & 0xFFFFu) | ((ncols & 0xFFFFu) << 16)); // d1 hi | tile_dim0
    g1[4] = (int)(nrows & 0xFFFFu);     // tile_dim1 (tile_dim2 = 0)
    g1[5] = (int)stride_elems;          // tensor_dim0_stride [31:0]
    g1[6] = 0;                          // stride0 hi | stride1 lo (unused for 2D)
    g1[7] = 0;

    v4i z = {};
#if defined(__clang_major__) && (__clang_major__ >= 23)
    v8i z8 = {};
    __builtin_amdgcn_tensor_load_to_lds(g0, g1, z, z, z8, 0);
#else
    __builtin_amdgcn_tensor_load_to_lds(g0, g1, z, z, 0);
#endif
}

// ---------------------------------------------------------------------------
// fp32 -> bf16 conversion
// ---------------------------------------------------------------------------
__global__ void cvt_f32_bf16(const float* __restrict__ in,
                             unsigned short* __restrict__ out, int n) {
    int i = blockIdx.x * blockDim.x + threadIdx.x;
    if (i < n) out[i] = f2bf(in[i]);
}

// ---------------------------------------------------------------------------
// QKV projection: P = x @ W^T  (x: [4096,1024], W: [1024,1024] row-major [e,d])
// z = 0: Q -> [B,H,S,64] bf16, scaled by 1/sqrt(64)
// z = 1: K -> [B,H,S,64] bf16
// z = 2: V -> [B,H,64,S] bf16 (transposed for the P@V WMMA A-operand)
// Block tile 128x128, 8 waves (4 M x 2 N), wave tile 32x64.
// Weight tile (128 x 32 bf16) staged into LDS by the TDM, double-buffered;
// K-loop unrolled x2 so the buffer alternation is static.
// ---------------------------------------------------------------------------
__global__ __launch_bounds__(256)
void qkv_gemm(const unsigned short* __restrict__ xb,
              const unsigned short* __restrict__ Wq,
              const unsigned short* __restrict__ Wk,
              const unsigned short* __restrict__ Wv,
              unsigned short* __restrict__ Qo,
              unsigned short* __restrict__ Ko,
              unsigned short* __restrict__ Vt) {
    __shared__ unsigned short btile[2][128 * LDS_ROW_ELEMS];   // 2 x 10 KB

    const int wave  = threadIdx.x >> 5;
    const int lane  = threadIdx.x & 31;
    const int waveM = wave & 3, waveN = wave >> 2;
    const int m0 = blockIdx.x * 128 + waveM * 32;
    const int nblk = blockIdx.y * 128;
    const int n0 = nblk + waveN * 64;
    const int z  = blockIdx.z;
    const unsigned short* W = (z == 0) ? Wq : (z == 1) ? Wk : Wv;
    const unsigned short* Wblk = W + (size_t)nblk * D_MODEL;

    const unsigned lds0 = (unsigned)(size_t)(&btile[0][0]);
    const unsigned lds1 = (unsigned)(size_t)(&btile[1][0]);

    if (wave == 0) {                       // prologue: stage k-chunk 0
        tdm_load_2d(lds0, Wblk, 32, 128, D_MODEL);
        __builtin_amdgcn_s_wait_tensorcnt(0);
    }
    __syncthreads();

    v8f acc[2][4] = {};
    const int nloc = waveN * 64;

    // one K-phase: prefetch next chunk, compute from `bt`, sync buffer swap
    auto phase = [&](int kk, const unsigned short* bt, unsigned lds_next) {
        if (wave == 0 && (kk + 32) < D_MODEL)
            tdm_load_2d(lds_next, Wblk + kk + 32, 32, 128, D_MODEL);

        Frag a0 = load_afrag(xb, D_MODEL, m0,      kk);
        Frag a1 = load_afrag(xb, D_MODEL, m0 + 16, kk);
        Frag b0 = load_bfrag_lds(bt, nloc);
        Frag b1 = load_bfrag_lds(bt, nloc + 16);
        Frag b2 = load_bfrag_lds(bt, nloc + 32);
        Frag b3 = load_bfrag_lds(bt, nloc + 48);
        // a0 pass then a1 pass: keeps all four B fragments live across the
        // whole sequence -> distinct registers -> ds_loads hoisted together.
        acc[0][0] = wmma_bf16(a0, b0, acc[0][0]);
        acc[0][1] = wmma_bf16(a0, b1, acc[0][1]);
        acc[0][2] = wmma_bf16(a0, b2, acc[0][2]);
        acc[0][3] = wmma_bf16(a0, b3, acc[0][3]);
        acc[1][0] = wmma_bf16(a1, b0, acc[1][0]);
        acc[1][1] = wmma_bf16(a1, b1, acc[1][1]);
        acc[1][2] = wmma_bf16(a1, b2, acc[1][2]);
        acc[1][3] = wmma_bf16(a1, b3, acc[1][3]);

        if ((kk + 32) < D_MODEL) {
            if (wave == 0) __builtin_amdgcn_s_wait_tensorcnt(0);
            __syncthreads();               // drains DScnt; buffer swap safe
        }
    };

    for (int k = 0; k < D_MODEL; k += 64) {
        phase(k,      btile[0], lds1);
        phase(k + 32, btile[1], lds0);
    }

    const float scale = (z == 0) ? 0.125f : 1.0f;   // 1/sqrt(D_HEAD) folded into Q
#pragma unroll
    for (int mf = 0; mf < 2; ++mf)
#pragma unroll
        for (int nf = 0; nf < 4; ++nf)
#pragma unroll
            for (int r = 0; r < 8; ++r) {
                const int m = m0 + mf * 16 + r + ((lane >> 4) << 3);
                const int n = n0 + nf * 16 + (lane & 15);
                const int b = m / SEQ, s = m % SEQ;
                const int h = n >> 6, d = n & 63;
                const unsigned short v = f2bf(acc[mf][nf][r] * scale);
                if (z == 2)
                    Vt[(((size_t)(b * NUM_HEADS + h) * D_HEAD + d) * SEQ) + s] = v;
                else if (z == 0)
                    Qo[(((size_t)(b * NUM_HEADS + h) * SEQ + s) * D_HEAD) + d] = v;
                else
                    Ko[(((size_t)(b * NUM_HEADS + h) * SEQ + s) * D_HEAD) + d] = v;
            }
}

// ---------------------------------------------------------------------------
// Flash attention (causal). Each wave owns 16 query rows (loop trip count is
// wave-dependent, so no block-level staging: K/V fragments come straight from
// the L0/L2-resident head buffers).
// Scores computed transposed: S^T[k,q] = K(16x64) @ Q^T(64x16)  (2 K-chunks)
// Online softmax stats per q-column, combined across lane-halves with
// __shfl_xor(...,16).  O^T[d,q] += V^T(16x32) @ P^T(32x16).
// ---------------------------------------------------------------------------
__global__ __launch_bounds__(256)
void flash_attn(const unsigned short* __restrict__ Q,
                const unsigned short* __restrict__ K,
                const unsigned short* __restrict__ Vt,
                unsigned short* __restrict__ attnout) {
    const int wave = threadIdx.x >> 5;
    const int lane = threadIdx.x & 31;
    const int half = lane >> 4;
    const int h = blockIdx.y, b = blockIdx.z;
    const int qb  = blockIdx.x * 128 + wave * 16;
    const int qhi = qb + 15;
    const int qrow = qb + (lane & 15);

    const unsigned short* Qbh = Q  + (size_t)(b * NUM_HEADS + h) * SEQ * D_HEAD;
    const unsigned short* Kbh = K  + (size_t)(b * NUM_HEADS + h) * SEQ * D_HEAD;
    const unsigned short* Vbh = Vt + (size_t)(b * NUM_HEADS + h) * D_HEAD * SEQ;

    // Q^T B-fragments (d-chunks 0..31 and 32..63), loaded once.
    Frag bq0 = load_bfrag(Qbh, D_HEAD, qb, 0);
    Frag bq1 = load_bfrag(Qbh, D_HEAD, qb, 32);

    float m = -INFINITY, l = 0.0f;
    v8f acc[4] = {};   // O^T: 4 d-tiles of 16, q on N

    for (int kt = 0; kt <= qhi; kt += 32) {
        // ---- scores for two 16-key tiles -------------------------------
        v8f s0 = {}, s1 = {};
        {
            Frag ka0 = load_afrag(Kbh, D_HEAD, kt, 0);
            Frag ka1 = load_afrag(Kbh, D_HEAD, kt, 32);
            s0 = wmma_bf16(ka0, bq0, s0);
            s0 = wmma_bf16(ka1, bq1, s0);
        }
        {
            Frag ka0 = load_afrag(Kbh, D_HEAD, kt + 16, 0);
            Frag ka1 = load_afrag(Kbh, D_HEAD, kt + 16, 32);
            s1 = wmma_bf16(ka0, bq0, s1);
            s1 = wmma_bf16(ka1, bq1, s1);
        }

        // ---- causal mask + online softmax (branchless, EXEC stays full) -
        float p0[8], p1[8];
        float tmax = -INFINITY;
#pragma unroll
        for (int r = 0; r < 8; ++r) {
            const int k0 = kt + r + (half << 3);   // tile0 key index
            const int k1 = k0 + 16;                // tile1 key index
            const float v0 = (k0 <= qrow) ? s0[r] : -INFINITY;
            const float v1 = (k1 <= qrow) ? s1[r] : -INFINITY;
            p0[r] = v0; p1[r] = v1;
            tmax = fmaxf(tmax, fmaxf(v0, v1));
        }
        tmax = fmaxf(tmax, __shfl_xor(tmax, 16, 32));   // combine lane-halves
        const float mnew  = fmaxf(m, tmax);
        const float alpha = __expf(m - mnew);           // 0 on first chunk
        float psum = 0.0f;
#pragma unroll
        for (int r = 0; r < 8; ++r) {
            p0[r] = __expf(p0[r] - mnew);
            p1[r] = __expf(p1[r] - mnew);
            psum += p0[r] + p1[r];
        }
        psum += __shfl_xor(psum, 16, 32);
        l = l * alpha + psum;
        m = mnew;
#pragma unroll
        for (int dt = 0; dt < 4; ++dt)
#pragma unroll
            for (int r = 0; r < 8; ++r) acc[dt][r] *= alpha;

        // ---- P^T (C layout, k = r + 8*half per tile) -> bf16 B-fragment
        //      (k = i + 16*half): cross-half exchange via shfl_xor(16).
        unsigned int pk0[4], pk1[4];
#pragma unroll
        for (int j = 0; j < 4; ++j) {
            pk0[j] = (unsigned)f2bf(p0[2 * j]) | ((unsigned)f2bf(p0[2 * j + 1]) << 16);
            pk1[j] = (unsigned)f2bf(p1[2 * j]) | ((unsigned)f2bf(p1[2 * j + 1]) << 16);
        }
        Frag bp;
#pragma unroll
        for (int j = 0; j < 4; ++j) {
            const unsigned int send = half ? pk0[j] : pk1[j];
            const unsigned int recv = __shfl_xor(send, 16, 32);
            bp.u[j]     = half ? recv   : pk0[j];   // k = 2j..  (+16*half)
            bp.u[4 + j] = half ? pk1[j] : recv;     // k = 8+2j..(+16*half)
        }

        // ---- O^T += V^T @ P^T ------------------------------------------
#pragma unroll
        for (int dt = 0; dt < 4; ++dt) {
            Frag va = load_afrag(Vbh, SEQ, dt * 16, kt);   // M=d rows of Vt
            acc[dt] = wmma_bf16(va, bp, acc[dt]);
        }
    }

    // ---- normalize and store to attn buffer [B,S,D_MODEL] (bf16) -------
    const float inv_l = 1.0f / l;
    const int s = qb + (lane & 15);
#pragma unroll
    for (int dt = 0; dt < 4; ++dt)
#pragma unroll
        for (int r = 0; r < 8; ++r) {
            const int d = h * D_HEAD + dt * 16 + r + (half << 3);
            attnout[(size_t)(b * SEQ + s) * D_MODEL + d] = f2bf(acc[dt][r] * inv_l);
        }
}

// ---------------------------------------------------------------------------
// Output projection: out = attn @ Wo^T  (fp32 output), TDM-staged Wo tile.
// ---------------------------------------------------------------------------
__global__ __launch_bounds__(256)
void out_gemm(const unsigned short* __restrict__ attn,
              const unsigned short* __restrict__ Wo,
              float* __restrict__ out) {
    __shared__ unsigned short btile[2][128 * LDS_ROW_ELEMS];

    const int wave  = threadIdx.x >> 5;
    const int lane  = threadIdx.x & 31;
    const int waveM = wave & 3, waveN = wave >> 2;
    const int m0 = blockIdx.x * 128 + waveM * 32;
    const int nblk = blockIdx.y * 128;
    const int n0 = nblk + waveN * 64;
    const unsigned short* Wblk = Wo + (size_t)nblk * D_MODEL;

    const unsigned lds0 = (unsigned)(size_t)(&btile[0][0]);
    const unsigned lds1 = (unsigned)(size_t)(&btile[1][0]);

    if (wave == 0) {
        tdm_load_2d(lds0, Wblk, 32, 128, D_MODEL);
        __builtin_amdgcn_s_wait_tensorcnt(0);
    }
    __syncthreads();

    v8f acc[2][4] = {};
    const int nloc = waveN * 64;

    auto phase = [&](int kk, const unsigned short* bt, unsigned lds_next) {
        if (wave == 0 && (kk + 32) < D_MODEL)
            tdm_load_2d(lds_next, Wblk + kk + 32, 32, 128, D_MODEL);

        Frag a0 = load_afrag(attn, D_MODEL, m0,      kk);
        Frag a1 = load_afrag(attn, D_MODEL, m0 + 16, kk);
        Frag b0 = load_bfrag_lds(bt, nloc);
        Frag b1 = load_bfrag_lds(bt, nloc + 16);
        Frag b2 = load_bfrag_lds(bt, nloc + 32);
        Frag b3 = load_bfrag_lds(bt, nloc + 48);
        acc[0][0] = wmma_bf16(a0, b0, acc[0][0]);
        acc[0][1] = wmma_bf16(a0, b1, acc[0][1]);
        acc[0][2] = wmma_bf16(a0, b2, acc[0][2]);
        acc[0][3] = wmma_bf16(a0, b3, acc[0][3]);
        acc[1][0] = wmma_bf16(a1, b0, acc[1][0]);
        acc[1][1] = wmma_bf16(a1, b1, acc[1][1]);
        acc[1][2] = wmma_bf16(a1, b2, acc[1][2]);
        acc[1][3] = wmma_bf16(a1, b3, acc[1][3]);

        if ((kk + 32) < D_MODEL) {
            if (wave == 0) __builtin_amdgcn_s_wait_tensorcnt(0);
            __syncthreads();
        }
    };

    for (int k = 0; k < D_MODEL; k += 64) {
        phase(k,      btile[0], lds1);
        phase(k + 32, btile[1], lds0);
    }

#pragma unroll
    for (int mf = 0; mf < 2; ++mf)
#pragma unroll
        for (int nf = 0; nf < 4; ++nf)
#pragma unroll
            for (int r = 0; r < 8; ++r) {
                const int mm = m0 + mf * 16 + r + ((lane >> 4) << 3);
                const int nn = n0 + nf * 16 + (lane & 15);
                out[(size_t)mm * D_MODEL + nn] = acc[mf][nf][r];
            }
}

// ---------------------------------------------------------------------------
// Launcher
// ---------------------------------------------------------------------------
extern "C" void kernel_launch(void* const* d_in, const int* in_sizes, int n_in,
                              void* d_out, int out_size, void* d_ws, size_t ws_size,
                              hipStream_t stream) {
    const float* x  = (const float*)d_in[0];
    const float* Wq = (const float*)d_in[1];
    const float* Wk = (const float*)d_in[2];
    const float* Wv = (const float*)d_in[3];
    const float* Wo = (const float*)d_in[4];
    float* out = (float*)d_out;

    // Workspace layout (bf16 = ushort), total 48 MB:
    unsigned short* ws = (unsigned short*)d_ws;
    const size_t NX = (size_t)M_TOTAL * D_MODEL;    // 4 M elems
    const size_t NW = (size_t)D_MODEL * D_MODEL;    // 1 M elems
    unsigned short* xb   = ws;
    unsigned short* wqb  = xb  + NX;
    unsigned short* wkb  = wqb + NW;
    unsigned short* wvb  = wkb + NW;
    unsigned short* wob  = wvb + NW;
    unsigned short* Qh   = wob + NW;                 // [B,H,S,64]
    unsigned short* Kh   = Qh  + NX;                 // [B,H,S,64]
    unsigned short* Vt   = Kh  + NX;                 // [B,H,64,S]
    unsigned short* attn = Vt  + NX;                 // [B,S,D]

    // 1) fp32 -> bf16 conversions
    cvt_f32_bf16<<<(int)((NX + 255) / 256), 256, 0, stream>>>(x,  xb,  (int)NX);
    cvt_f32_bf16<<<(int)((NW + 255) / 256), 256, 0, stream>>>(Wq, wqb, (int)NW);
    cvt_f32_bf16<<<(int)((NW + 255) / 256), 256, 0, stream>>>(Wk, wkb, (int)NW);
    cvt_f32_bf16<<<(int)((NW + 255) / 256), 256, 0, stream>>>(Wv, wvb, (int)NW);
    cvt_f32_bf16<<<(int)((NW + 255) / 256), 256, 0, stream>>>(Wo, wob, (int)NW);

    // 2) Q/K/V projections (z selects weight + output layout)
    qkv_gemm<<<dim3(M_TOTAL / 128, D_MODEL / 128, 3), 256, 0, stream>>>(
        xb, wqb, wkb, wvb, Qh, Kh, Vt);

    // 3) causal flash attention
    flash_attn<<<dim3(SEQ / 128, NUM_HEADS, BATCH), 256, 0, stream>>>(
        Qh, Kh, Vt, attn);

    // 4) output projection -> fp32
    out_gemm<<<dim3(M_TOTAL / 128, D_MODEL / 128), 256, 0, stream>>>(
        attn, wob, out);
}
